// GATModel_15212774162446
// MI455X (gfx1250) — matrix operand order repaired and checked by reading, not compiled, not run on hardware
//
#include <hip/hip_runtime.h>
#include <hip/hip_bf16.h>
#include <math.h>

typedef __attribute__((ext_vector_type(16))) _Float16 v16h;
typedef __attribute__((ext_vector_type(8)))  _Float16 v8h;
typedef __attribute__((ext_vector_type(8)))  float    v8f;

#define NEG_SLOPE 0.2f

// ---------- order-preserving float <-> uint encoding for atomic max ----------
__device__ __forceinline__ unsigned fenc(float f) {
    unsigned u = __float_as_uint(f);
    return (u & 0x80000000u) ? ~u : (u | 0x80000000u);
}
__device__ __forceinline__ float fdec(unsigned u) {
    return (u & 0x80000000u) ? __uint_as_float(u ^ 0x80000000u)
                             : __uint_as_float(~u);
}

// ---------- fp32 -> fp16 conversion (activations, row-major kept) ----------
__global__ void cvt_f32_to_f16(const float* __restrict__ in,
                               _Float16* __restrict__ out, long long n) {
    long long i = (long long)blockIdx.x * blockDim.x + threadIdx.x;
    if (i < n) out[i] = (_Float16)in[i];
}

// ---------- fp32 W[K,Nout] -> fp16 WT[Nout,K] (so B-fragments load contiguously) ----
__global__ void transpose_w_f16(const float* __restrict__ W, _Float16* __restrict__ WT,
                                int K, int Nout) {
    long long i = (long long)blockIdx.x * blockDim.x + threadIdx.x;
    if (i >= (long long)K * Nout) return;
    int k = (int)(i / Nout), n = (int)(i % Nout);
    WT[(size_t)n * K + k] = (_Float16)W[i];
}

__global__ void zero_f32(float* __restrict__ p, long long n) {
    long long i = (long long)blockIdx.x * blockDim.x + threadIdx.x;
    if (i < n) p[i] = 0.0f;
}

// ---------- WMMA GEMM: C[M,Nout] = A[M,K](f16,row) x W (via WT[Nout,K] f16), fp32 acc
// one wave computes a 16 x (16*TN) strip; M%16==0, K%32==0, Nout%(16*TN)==0
template <int TN>
__global__ __launch_bounds__(128)
void gemm_wmma_f16(const _Float16* __restrict__ A, const _Float16* __restrict__ WT,
                   float* __restrict__ C, int M, int K, int Nout) {
    const int lane = threadIdx.x & 31;
    const int wid  = threadIdx.x >> 5;
    const int stripsN = Nout / (16 * TN);
    const int strip = blockIdx.x * 4 + wid;
    if (strip >= (M >> 4) * stripsN) return;          // wave-uniform: EXEC stays full
    const int m0 = (strip / stripsN) << 4;
    const int n0 = (strip % stripsN) * (16 * TN);
    const int hs = lane >> 4;                         // half-wave select
    const int r  = lane & 15;                         // A row / B,C column within tile

    v8f acc[TN];
#pragma unroll
    for (int j = 0; j < TN; ++j) acc[j] = (v8f){};

    const _Float16* arow = A + (size_t)(m0 + r) * K + hs * 8;
    for (int k0 = 0; k0 < K; k0 += 32) {
        // A frag (ISA 16-bit A 16x32): lanes<16 hold K {0..7,16..23}; lanes>=16 {8..15,24..31}
        v8h alo = *(const v8h*)(arow + k0);
        v8h ahi = *(const v8h*)(arow + k0 + 16);
        v16h a;
#pragma unroll
        for (int i = 0; i < 8; ++i) { a[i] = alo[i]; a[8 + i] = ahi[i]; }
#pragma unroll
        for (int j = 0; j < TN; ++j) {
            // B frag (32x16): lane holds column n0+16j+r, K {8hs..8hs+7, 16+8hs..16+8hs+7} (matches A)
            const _Float16* bp = WT + (size_t)(n0 + j * 16 + r) * K + k0 + hs * 8;
            v8h blo = *(const v8h*)(bp);
            v8h bhi = *(const v8h*)(bp + 16);
            v16h b;
#pragma unroll
            for (int i = 0; i < 8; ++i) { b[i] = blo[i]; b[8 + i] = bhi[i]; }
            acc[j] = __builtin_amdgcn_wmma_f32_16x16x32_f16(false, a, false, b,
                                                            (short)0, acc[j], false, false);
        }
    }
    // C/D layout: VGPR i -> row m0 + 8*hs + i, column tile_n0 + r
#pragma unroll
    for (int j = 0; j < TN; ++j) {
        float* crow = C + (size_t)(m0 + hs * 8) * Nout + n0 + j * 16 + r;
#pragma unroll
        for (int i = 0; i < 8; ++i) crow[(size_t)i * Nout] = acc[j][i];
    }
}

// ---------- per-(node,head): attention logit pieces + softmax state init ----------
__global__ void node_prep(const float* __restrict__ h, const float* __restrict__ asrc,
                          const float* __restrict__ adst, float* __restrict__ s_src,
                          float* __restrict__ s_dst, unsigned* __restrict__ mbuf,
                          float* __restrict__ denom, int N, int H, int D, int F) {
    int t = blockIdx.x * blockDim.x + threadIdx.x;
    if (t >= N * H) return;
    int n = t / H, hd = t % H;
    const float* hp = h + (size_t)n * F + hd * D;
    const float* as = asrc + hd * D;
    const float* ad = adst + hd * D;
    float ss = 0.f, sd = 0.f;
    for (int d = 0; d < D; ++d) { float v = hp[d]; ss += v * as[d]; sd += v * ad[d]; }
    s_src[t] = ss; s_dst[t] = sd;
    mbuf[t]  = fenc(-INFINITY);
    denom[t] = 0.f;
}

__device__ __forceinline__ void edge_ends(const int* __restrict__ ei, long long e,
                                          int E, int& s, int& d) {
    if (e < E) { s = ei[e]; d = ei[(long long)E + e]; }
    else       { s = d = (int)(e - E); }               // self-loop
}

// ---------- edge pass 1: segment max ----------
__global__ void edge_max(const int* __restrict__ ei, const float* __restrict__ s_src,
                         const float* __restrict__ s_dst, unsigned* __restrict__ mbuf,
                         int E, long long ET, int H) {
    long long t = (long long)blockIdx.x * blockDim.x + threadIdx.x;
    if (t >= ET * H) return;
    long long e = t / H; int hd = (int)(t % H);
    int s, d; edge_ends(ei, e, E, s, d);
    float ev = s_src[(size_t)s * H + hd] + s_dst[(size_t)d * H + hd];
    ev = ev > 0.f ? ev : NEG_SLOPE * ev;
    atomicMax(&mbuf[(size_t)d * H + hd], fenc(ev));
}

// ---------- edge pass 2: segment sum of exp(e - m) ----------
__global__ void edge_sum(const int* __restrict__ ei, const float* __restrict__ s_src,
                         const float* __restrict__ s_dst, const unsigned* __restrict__ mbuf,
                         float* __restrict__ denom, int E, long long ET, int H) {
    long long t = (long long)blockIdx.x * blockDim.x + threadIdx.x;
    if (t >= ET * H) return;
    long long e = t / H; int hd = (int)(t % H);
    int s, d; edge_ends(ei, e, E, s, d);
    float ev = s_src[(size_t)s * H + hd] + s_dst[(size_t)d * H + hd];
    ev = ev > 0.f ? ev : NEG_SLOPE * ev;
    float w = __expf(ev - fdec(mbuf[(size_t)d * H + hd]));
    atomicAdd(&denom[(size_t)d * H + hd], w);
}

// ---------- edge pass 3: alpha-weighted gather/scatter, 4 floats per thread ----------
__global__ void edge_aggr(const int* __restrict__ ei, const float* __restrict__ s_src,
                          const float* __restrict__ s_dst, const unsigned* __restrict__ mbuf,
                          const float* __restrict__ denom, const float* __restrict__ h,
                          float* __restrict__ agg, int E, long long ET, int H, int D, int F) {
    int chunks = F >> 2;
    long long t = (long long)blockIdx.x * blockDim.x + threadIdx.x;
    if (t >= ET * chunks) return;
    long long e = t / chunks; int c = (int)(t % chunks);
    int s, d; edge_ends(ei, e, E, s, d);
    int f0 = c << 2;
    int hd = f0 / D;
    float ev = s_src[(size_t)s * H + hd] + s_dst[(size_t)d * H + hd];
    ev = ev > 0.f ? ev : NEG_SLOPE * ev;
    float w = __expf(ev - fdec(mbuf[(size_t)d * H + hd]));
    float alpha = w / denom[(size_t)d * H + hd];
    const float4 hv = *(const float4*)(h + (size_t)s * F + f0);
    float* op = agg + (size_t)d * F + f0;
    atomicAdd(op + 0, alpha * hv.x);
    atomicAdd(op + 1, alpha * hv.y);
    atomicAdd(op + 2, alpha * hv.z);
    atomicAdd(op + 3, alpha * hv.w);
}

// ---------- epilogue (layers 0/1): bias + elu, stage fp16 input for next GEMM ----------
__global__ void epilogue_elu(const float* __restrict__ agg, const float* __restrict__ b,
                             _Float16* __restrict__ xh16, int N, int F) {
    long long i = (long long)blockIdx.x * blockDim.x + threadIdx.x;
    if (i >= (long long)N * F) return;
    int f = (int)(i % F);
    float v = agg[i] + b[f];
    v = v > 0.f ? v : (__expf(v) - 1.0f);   // elu(alpha=1)
    xh16[i] = (_Float16)v;
}

// ---------- final: bias + log_softmax over 32 classes ----------
__global__ void final_logsoftmax(const float* __restrict__ agg, const float* __restrict__ b,
                                 float* __restrict__ out, int N, int C) {
    int n = blockIdx.x * blockDim.x + threadIdx.x;
    if (n >= N) return;
    const float* ap = agg + (size_t)n * C;
    float mx = -INFINITY;
    for (int j = 0; j < C; ++j) { float v = ap[j] + b[j]; mx = v > mx ? v : mx; }
    float se = 0.f;
    for (int j = 0; j < C; ++j) se += __expf(ap[j] + b[j] - mx);
    float lse = mx + __logf(se);
    float* op = out + (size_t)n * C;
    for (int j = 0; j < C; ++j) op[j] = ap[j] + b[j] - lse;
}

// ============================== launcher ==============================
extern "C" void kernel_launch(void* const* d_in, const int* in_sizes, int n_in,
                              void* d_out, int out_size, void* d_ws, size_t ws_size,
                              hipStream_t stream) {
    const float* x    = (const float*)d_in[0];
    const int*   ei   = (const int*)d_in[1];
    const float* W[3] = {(const float*)d_in[2], (const float*)d_in[6], (const float*)d_in[10]};
    const float* b[3] = {(const float*)d_in[3], (const float*)d_in[7], (const float*)d_in[11]};
    const float* as[3]= {(const float*)d_in[4], (const float*)d_in[8], (const float*)d_in[12]};
    const float* ad[3]= {(const float*)d_in[5], (const float*)d_in[9], (const float*)d_in[13]};
    float* out = (float*)d_out;

    const int IN = 128, HID = 256, OUT = 32;
    const int N = in_sizes[0] / IN;
    const int E = in_sizes[1] / 2;
    const long long ET = (long long)E + N;   // with self-loops

    // workspace carve-up (all chunk sizes are multiples of 256B)
    char* p = (char*)d_ws;
    _Float16* xh16 = (_Float16*)p;               p += (size_t)N * HID * 2;
    _Float16* W16T = (_Float16*)p;               p += (size_t)HID * HID * 2;
    float*    h32  = (float*)p;                  p += (size_t)N * HID * 4;
    float*    agg  = (float*)p;                  p += (size_t)N * HID * 4;
    float*    ssrc = (float*)p;                  p += (size_t)N * 4 * 4;
    float*    sdst = (float*)p;                  p += (size_t)N * 4 * 4;
    unsigned* mbuf = (unsigned*)p;               p += (size_t)N * 4 * 4;
    float*    den  = (float*)p;                  p += (size_t)N * 4 * 4;

    const int T = 256;
    auto cdiv = [](long long a, long long b) { return (unsigned)((a + b - 1) / b); };

    const int Kl[3]  = {IN, HID, HID};
    const int Nl[3]  = {HID, HID, OUT};
    const int Hl[3]  = {4, 4, 1};
    const int Dl[3]  = {64, 64, 32};

    // stage layer-0 activations to fp16
    cvt_f32_to_f16<<<cdiv((long long)N * IN, T), T, 0, stream>>>(x, xh16, (long long)N * IN);

    for (int l = 0; l < 3; ++l) {
        const int K = Kl[l], F = Nl[l], H = Hl[l], D = Dl[l];
        // weights -> fp16, transposed to [Nout, K] for contiguous B-fragment loads
        transpose_w_f16<<<cdiv((long long)K * F, T), T, 0, stream>>>(W[l], W16T, K, F);
        // h = act @ W  (WMMA, fp32 accumulate)
        if (F == 256) {
            long long strips = (long long)(N / 16) * (F / 64);   // TN = 4
            gemm_wmma_f16<4><<<cdiv(strips, 4), 128, 0, stream>>>(xh16, W16T, h32, N, K, F);
        } else {
            long long strips = (long long)(N / 16) * (F / 32);   // TN = 2
            gemm_wmma_f16<2><<<cdiv(strips, 4), 128, 0, stream>>>(xh16, W16T, h32, N, K, F);
        }
        // attention prep + zero aggregation target
        zero_f32<<<cdiv((long long)N * F, T), T, 0, stream>>>(agg, (long long)N * F);
        node_prep<<<cdiv((long long)N * H, T), T, 0, stream>>>(h32, as[l], ad[l], ssrc, sdst,
                                                               mbuf, den, N, H, D, F);
        // segment softmax + aggregation over edges
        edge_max <<<cdiv(ET * H, T), T, 0, stream>>>(ei, ssrc, sdst, mbuf, E, ET, H);
        edge_sum <<<cdiv(ET * H, T), T, 0, stream>>>(ei, ssrc, sdst, mbuf, den, E, ET, H);
        edge_aggr<<<cdiv(ET * (F / 4), T), T, 0, stream>>>(ei, ssrc, sdst, mbuf, den, h32,
                                                           agg, E, ET, H, D, F);
        if (l < 2) {
            epilogue_elu<<<cdiv((long long)N * F, T), T, 0, stream>>>(agg, b[l], xh16, N, F);
        } else {
            final_logsoftmax<<<cdiv(N, T), T, 0, stream>>>(agg, b[l], out, N, OUT);
        }
    }
    (void)n_in; (void)out_size; (void)ws_size;
}
